// ExpertChoiceMoEMatcher_40724879900937
// MI455X (gfx1250) — matrix operand order repaired and checked by reading, not compiled, not run on hardware
//
#include <hip/hip_runtime.h>
#include <hip/hip_bf16.h>

// ---------------------------------------------------------------------------
// Problem constants (from reference): B_T=16384, D=1024, E=16, k=1024
// ---------------------------------------------------------------------------
#define BT   16384
#define DD   1024
#define EE   16
#define KSEL 1024   // k_nodes = BT/E

// GEMM tiling
#define BM 128
#define BN 64
#define BK 32
#define NT (DD / BK)         // K-steps
#define LDS_PITCH (BK + 8)   // 40 bf16 = 80 bytes -> conflict-free b128 frag loads

typedef __attribute__((ext_vector_type(16))) __bf16 bf16x16;
typedef __attribute__((ext_vector_type(8)))  __bf16 bf16x8;
typedef __attribute__((ext_vector_type(8)))  float  f32x8;

// ---------------------------------------------------------------------------
// Fragment helpers (layouts per CDNA5 ISA 7.12.2, 16-bit A 16x32 / B 32x16)
// A, lane<16: elems 0..7 = K 0..7, elems 8..15 = K 16..23 (lane>=16: +8)
// B, lane<16: elems 0..7 = K 0..7, elems 8..15 = K 16..23 (lane>=16: +8), col N = lane&15
// ---------------------------------------------------------------------------
__device__ __forceinline__ bf16x16 load_frag_A(const __bf16* row, int kOff) {
  bf16x8 lo = *(const bf16x8*)(row + kOff);        // K = kOff..kOff+7
  bf16x8 hi = *(const bf16x8*)(row + 16 + kOff);   // K = 16+kOff..
  return __builtin_shufflevector(lo, hi, 0,1,2,3,4,5,6,7,8,9,10,11,12,13,14,15);
}
__device__ __forceinline__ bf16x16 load_frag_B(const __bf16* row, int kOff) {
  bf16x8 lo = *(const bf16x8*)(row + kOff);        // K = kOff..kOff+7
  bf16x8 hi = *(const bf16x8*)(row + 16 + kOff);   // K = 16+kOff..
  return __builtin_shufflevector(lo, hi, 0,1,2,3,4,5,6,7,8,9,10,11,12,13,14,15);
}
// packed bf16 sign-flip (bf16 WMMA has no A/B NEG modifier, only C)
__device__ __forceinline__ bf16x16 neg_frag(bf16x16 v) {
  union { bf16x16 v; unsigned u[8]; } a; a.v = v;
#pragma unroll
  for (int i = 0; i < 8; ++i) a.u[i] ^= 0x80008000u;
  return a.v;
}

// ---------------------------------------------------------------------------
// 0) zero output accumulator + counts
// ---------------------------------------------------------------------------
__global__ void moe_zero_kernel(float* __restrict__ res, float* __restrict__ counts) {
  size_t i = (size_t)blockIdx.x * blockDim.x + threadIdx.x;
  if (i < (size_t)BT * DD * 2) res[i] = 0.0f;
  if (i < BT) counts[i] = 0.0f;
}

// ---------------------------------------------------------------------------
// 1) gating scores: one wave32 per token; x read once, gw (128KB) L2-resident
// ---------------------------------------------------------------------------
__global__ __launch_bounds__(256)
void moe_gate_kernel(const float* __restrict__ x, const float* __restrict__ gw,
                     float* __restrict__ scores) {
  int wave = (int)((blockIdx.x * blockDim.x + threadIdx.x) >> 5);
  int lane = threadIdx.x & 31;
  if (wave >= BT) return;
  const float* xr = x + (size_t)wave * (DD * 2);
  float acc[EE] = {};
  for (int c = lane; c < DD * 2; c += 32) {
    float v = xr[c];
    const float* g = gw + (size_t)c * EE;
#pragma unroll
    for (int e = 0; e < EE; ++e) acc[e] = fmaf(v, g[e], acc[e]);
  }
#pragma unroll
  for (int e = 0; e < EE; ++e) {
    float a = acc[e];
    for (int off = 16; off > 0; off >>= 1) a += __shfl_down(a, off, 32);
    if (lane == 0) scores[(size_t)wave * EE + e] = a;
  }
}

// ---------------------------------------------------------------------------
// 2) top-k per expert: full bitonic sort of 16384 64-bit keys in 128KB LDS
//    key = ordered(float) << 32 | (BT-1-idx)  -> descending sort gives
//    score-desc, index-asc tie-break (matches jax.lax.top_k)
// ---------------------------------------------------------------------------
__global__ __launch_bounds__(1024)
void moe_topk_kernel(const float* __restrict__ scores, int* __restrict__ idxOut,
                     float* __restrict__ scOut, float* __restrict__ counts) {
  extern __shared__ unsigned long long keys[];   // BT entries = 128KB (CDNA5 LDS: 320KB)
  const int e = blockIdx.x;
  const int tid = threadIdx.x;

  for (int i = tid; i < BT; i += 1024) {
    float s = scores[(size_t)i * EE + e];
    unsigned u = __float_as_uint(s);
    u = (u & 0x80000000u) ? ~u : (u | 0x80000000u);   // monotonic float ordering
    keys[i] = ((unsigned long long)u << 32) | (unsigned)(BT - 1 - i);
  }
  __syncthreads();

  for (int size = 2; size <= BT; size <<= 1) {
    for (int stride = size >> 1; stride > 0; stride >>= 1) {
      for (int t = tid; t < BT / 2; t += 1024) {
        int i = ((t / stride) * (stride << 1)) + (t % stride);
        int j = i + stride;
        bool desc = ((i & size) == 0);
        unsigned long long a = keys[i], b = keys[j];
        if (desc ? (a < b) : (a > b)) { keys[i] = b; keys[j] = a; }
      }
      __syncthreads();
    }
  }

  if (tid < KSEL) {
    unsigned long long key = keys[tid];
    int i = BT - 1 - (int)(key & 0xFFFFFFFFu);
    idxOut[tid * EE + e] = i;
    scOut[tid * EE + e]  = scores[(size_t)i * EE + e];
    atomicAdd(&counts[i], 1.0f);
  }
}

// ---------------------------------------------------------------------------
// 3) grouped complex GEMM via bf16 WMMA, f32 accumulate
//    grid = (N/BN, M/BM, E); 256 threads = 8 waves, wave tile 32x32
//    Register->LDS double buffering: global loads for tile k+1 issued before
//    the WMMAs of tile k; one barrier per K-step.
// ---------------------------------------------------------------------------
__global__ __launch_bounds__(256)
void moe_gemm_kernel(const float* __restrict__ x, const float* __restrict__ w,
                     const int* __restrict__ topkIdx, const float* __restrict__ topkSc,
                     float* __restrict__ res) {
  __shared__ __bf16 sXr[2][BM][LDS_PITCH], sXi[2][BM][LDS_PITCH];
  __shared__ __bf16 sWr[2][BN][LDS_PITCH], sWi[2][BN][LDS_PITCH];
  __shared__ int   sIdx[BM];
  __shared__ float sSc[BM];

  const int nb = blockIdx.x, mb = blockIdx.y, e = blockIdx.z;
  const int tid  = threadIdx.x;
  const int lane = tid & 31, wave = tid >> 5;
  const int wm = wave & 3, wn = wave >> 2;        // 4x2 wave grid over 128x64

  if (tid < BM) {
    int slot = mb * BM + tid;
    sIdx[tid] = topkIdx[slot * EE + e];
    sSc[tid]  = topkSc[slot * EE + e];
  }
  __syncthreads();

  const float2* x2 = (const float2*)x;            // [BT][D] complex
  const float2* w2 = (const float2*)w;            // [E][D(k)][D(n)] complex

  // loader mappings
  const int rX = tid >> 1, hX = tid & 1;          // X: row, 16-col half
  const int kW = tid >> 3, gW = tid & 7;          // W: k row, 8-col group
  const int tokX = sIdx[rX];
  const float2* baseX = x2 + ((size_t)tokX * DD + hX * 16);
  const float2* baseW = w2 + (((size_t)e * DD + kW) * DD + (size_t)nb * BN + gW * 8);

  float2 pX[16];   // prefetch registers (live across compute)
  float2 pW[8];

  // ---- prologue: fetch tile 0, stage into LDS buffer 0 ----
#pragma unroll
  for (int j = 0; j < 16; ++j) pX[j] = baseX[j];
#pragma unroll
  for (int j = 0; j < 8; ++j)  pW[j] = baseW[j];
#pragma unroll
  for (int j = 0; j < 16; ++j) {
    sXr[0][rX][hX * 16 + j] = (__bf16)pX[j].x;
    sXi[0][rX][hX * 16 + j] = (__bf16)pX[j].y;
  }
#pragma unroll
  for (int j = 0; j < 8; ++j) {
    sWr[0][gW * 8 + j][kW] = (__bf16)pW[j].x;
    sWi[0][gW * 8 + j][kW] = (__bf16)pW[j].y;
  }
  __syncthreads();

  f32x8 accR[2][2] = {};
  f32x8 accI[2][2] = {};

  for (int kt = 0; kt < NT; ++kt) {
    const int cur = kt & 1, nxt = cur ^ 1;

    // ---- issue global loads for tile kt+1 (overlap with WMMA below) ----
    if (kt + 1 < NT) {
      const float2* srcX = baseX + (size_t)(kt + 1) * BK;
      const float2* srcW = baseW + (size_t)(kt + 1) * BK * DD;
#pragma unroll
      for (int j = 0; j < 16; ++j) pX[j] = srcX[j];
#pragma unroll
      for (int j = 0; j < 8; ++j)  pW[j] = srcW[j];
    }

    // ---- fragments + 16 WMMAs per wave from LDS[cur] ----
    {
      const int mrow = lane & 15;
      const int kA = (lane >> 4) * 8, kB = (lane >> 4) * 8;
      bf16x16 Ar[2], Ai[2], AiN[2], Br[2], Bi[2];
#pragma unroll
      for (int mi = 0; mi < 2; ++mi) {
        int m = wm * 32 + mi * 16 + mrow;
        Ar[mi]  = load_frag_A(&sXr[cur][m][0], kA);
        Ai[mi]  = load_frag_A(&sXi[cur][m][0], kA);
        AiN[mi] = neg_frag(Ai[mi]);
      }
#pragma unroll
      for (int ni = 0; ni < 2; ++ni) {
        int n = wn * 32 + ni * 16 + mrow;
        Br[ni] = load_frag_B(&sWr[cur][n][0], kB);
        Bi[ni] = load_frag_B(&sWi[cur][n][0], kB);
      }
#pragma unroll
      for (int mi = 0; mi < 2; ++mi)
#pragma unroll
        for (int ni = 0; ni < 2; ++ni) {
          // Yr = Xr*Wr - Xi*Wi ; Yi = Xr*Wi + Xi*Wr
          accR[mi][ni] = __builtin_amdgcn_wmma_f32_16x16x32_bf16(
              false, Ar[mi],  false, Br[ni], (short)0, accR[mi][ni], false, false);
          accR[mi][ni] = __builtin_amdgcn_wmma_f32_16x16x32_bf16(
              false, AiN[mi], false, Bi[ni], (short)0, accR[mi][ni], false, false);
          accI[mi][ni] = __builtin_amdgcn_wmma_f32_16x16x32_bf16(
              false, Ar[mi],  false, Bi[ni], (short)0, accI[mi][ni], false, false);
          accI[mi][ni] = __builtin_amdgcn_wmma_f32_16x16x32_bf16(
              false, Ai[mi],  false, Br[ni], (short)0, accI[mi][ni], false, false);
        }
    }

    // ---- convert & stage tile kt+1 into LDS[nxt]; single barrier ----
    if (kt + 1 < NT) {
#pragma unroll
      for (int j = 0; j < 16; ++j) {
        sXr[nxt][rX][hX * 16 + j] = (__bf16)pX[j].x;
        sXi[nxt][rX][hX * 16 + j] = (__bf16)pX[j].y;
      }
#pragma unroll
      for (int j = 0; j < 8; ++j) {
        sWr[nxt][gW * 8 + j][kW] = (__bf16)pW[j].x;
        sWi[nxt][gW * 8 + j][kW] = (__bf16)pW[j].y;
      }
    }
    __syncthreads();
  }

  // ---- epilogue: scale by gate score, scatter-add into token rows ----
  // C/D layout: VGPR r, lanes 0-15 -> M=r, N=lane ; lanes 16-31 -> M=r+8
  const int laneHi = lane >> 4;
#pragma unroll
  for (int mi = 0; mi < 2; ++mi)
#pragma unroll
    for (int ni = 0; ni < 2; ++ni)
#pragma unroll
      for (int r = 0; r < 8; ++r) {
        int mL = wm * 32 + mi * 16 + r + laneHi * 8;
        int token = sIdx[mL];
        float s = sSc[mL];
        int ng = nb * BN + wn * 32 + ni * 16 + (lane & 15);
        size_t o = ((size_t)token * DD + ng) * 2;
        atomicAdd(res + o,     accR[mi][ni][r] * s);
        atomicAdd(res + o + 1, accI[mi][ni][r] * s);
      }
}

// ---------------------------------------------------------------------------
// 4) combine-normalize + ModReLU (in place on d_out accumulator)
// ---------------------------------------------------------------------------
__global__ void moe_modrelu_kernel(float2* __restrict__ res,
                                   const float* __restrict__ counts,
                                   const float* __restrict__ bias) {
  size_t i = (size_t)blockIdx.x * blockDim.x + threadIdx.x;  // over BT*D
  if (i >= (size_t)BT * DD) return;
  int t = (int)(i >> 10);
  int d = (int)(i & (DD - 1));
  float cnt = fmaxf(counts[t], 1.0f);
  float2 z = res[i];
  float zr = z.x / cnt, zi = z.y / cnt;
  float mag = sqrtf(zr * zr + zi * zi);
  float act = fmaxf(mag + bias[d], 0.0f);
  float inv = act / fmaxf(mag, 1e-8f);
  res[i] = make_float2(zr * inv, zi * inv);
}

// ---------------------------------------------------------------------------
// launch
// d_out layout (tuple, flat): res[BT*D*2] | topk_idx[1024*16] (int bits) |
//                             topk_scores[1024*16] | counts[BT]
// ---------------------------------------------------------------------------
extern "C" void kernel_launch(void* const* d_in, const int* in_sizes, int n_in,
                              void* d_out, int out_size, void* d_ws, size_t ws_size,
                              hipStream_t stream) {
  const float* x    = (const float*)d_in[0];
  const float* gw   = (const float*)d_in[1];
  const float* w    = (const float*)d_in[2];
  const float* bias = (const float*)d_in[3];

  float* res = (float*)d_out;
  const size_t RES = (size_t)BT * DD * 2;
  int*   idxOut = (int*)(res + RES);
  float* scOut  = res + RES + (size_t)KSEL * EE;
  float* counts = res + RES + 2 * (size_t)KSEL * EE;

  float* scores = (float*)d_ws;   // [BT, E] = 1 MB scratch

  moe_zero_kernel<<<(unsigned)((RES + 255) / 256), 256, 0, stream>>>(res, counts);
  moe_gate_kernel<<<BT / 8, 256, 0, stream>>>(x, gw, scores);
  moe_topk_kernel<<<EE, 1024, BT * sizeof(unsigned long long), stream>>>(
      scores, idxOut, scOut, counts);
  dim3 g(DD / BN, KSEL / BM, EE);
  moe_gemm_kernel<<<g, 256, 0, stream>>>(x, w, idxOut, scOut, res);
  moe_modrelu_kernel<<<(BT * DD) / 256, 256, 0, stream>>>((float2*)res, counts, bias);
}
